// varletNetworks_32143535243281
// MI455X (gfx1250) — matrix operand order repaired and checked by reading, not compiled, not run on hardware
//
#include <hip/hip_runtime.h>

// CDNA5 / gfx1250: wave32, WMMA 16x16x32 f16 -> f32 accumulate.
typedef __attribute__((ext_vector_type(16))) _Float16 v16h;
typedef __attribute__((ext_vector_type(8)))  _Float16 v8h;
typedef __attribute__((ext_vector_type(4)))  _Float16 v4h;
typedef __attribute__((ext_vector_type(8)))  float    v8f;

#define H_STEP 0.1f
#define TVEPS  1e-3f

static inline int imin_host(int a, int b) { return a < b ? a : b; }

__global__ void zero_kernel(float* __restrict__ p, int n) {
    const int gid = blockIdx.x * blockDim.x + threadIdx.x;
    const int stride = gridDim.x * blockDim.x;
    const int n4 = n >> 2;
    float4* p4 = (float4*)p;
    const float4 z4 = make_float4(0.f, 0.f, 0.f, 0.f);
    for (int i = gid; i < n4; i += stride) p4[i] = z4;
    for (int i = (n4 << 2) + gid; i < n; i += stride) p[i] = 0.0f;
}

// out[N,128] = transpose(in[128,N]); 32x32 LDS tiles.
__global__ void __launch_bounds__(256)
transpose128(const float* __restrict__ in, float* __restrict__ out, int N)
{
    __shared__ float tile[32][33];
    const int c0 = blockIdx.x * 32;
    const int r0 = blockIdx.y * 32;            // gridDim.y == 4 (128 rows)
    const int x = threadIdx.x & 31, y0 = threadIdx.x >> 5;
#pragma unroll
    for (int dy = 0; dy < 32; dy += 8) {
        const int c = c0 + x;
        tile[y0 + dy][x] = (c < N) ? in[(size_t)(r0 + y0 + dy) * N + c] : 0.0f;
    }
    __syncthreads();
#pragma unroll
    for (int dy = 0; dy < 32; dy += 8) {
        const int c = c0 + y0 + dy;
        if (c < N) out[(size_t)c * 128 + r0 + x] = tile[x][y0 + dy];
    }
}

// Y[128, N] = Kmat[128,128] @ B[128, N]
//   BMODE 0 (ROW):    B = X, row-major [128, N]  (NX == N, k-stride NX)
//   BMODE 1 (COL):    B[k, c] = X[c*128 + k]     (column-contiguous, e.g. zT)
//   BMODE 2 (GATHER): B[k, c] = X[iInd[c]*128+k] - X[jInd[c]*128+k]  (xnT)
//   RELU:  Y = max(Y, 0) before store/stats
//   STATS: accumulate per-row sum / sum-of-squares into gStats[2*row(+1)]
//
// LDS B tile is pre-swizzled into the WMMA B register layout:
//   sB[kc][lane][e], e = 0..15 contiguous per lane -> 2x ds_load_b128.
//   B element (k = kc*32 + e + 16*(lane>>4), col = col0 + (lane&15)).
#define BSTRIDE 24                    // halves per lane slot (48B, 16B-aligned)
#define BCHUNK  (32 * BSTRIDE)

template <int BMODE, bool RELU, bool STATS>
__global__ void __launch_bounds__(256)
gemm_k128(const float* __restrict__ Kmat, const float* __restrict__ X,
          const int* __restrict__ iInd, const int* __restrict__ jInd,
          float* __restrict__ Y, float* __restrict__ gStats,
          int N, int NX)
{
    __shared__ _Float16 sB[4 * BCHUNK];   // 6 KB
    __shared__ float sSum[128];
    __shared__ float sSq[128];

    if (STATS && threadIdx.x < 128) {
        sSum[threadIdx.x] = 0.0f;
        sSq[threadIdx.x]  = 0.0f;
    }

    const int lane = threadIdx.x & 31;
    const int wave = threadIdx.x >> 5;   // 0..7 -> row tile
    const int m0   = wave << 4;
    const int mrow = lane & 15;
    const int hi   = lane >> 4;

    // A fragments (this code's contiguous k order, matching sB): lane holds row M=lane&15;
// element e -> K = e + 16*(lane>>4).
    v16h a[4];
#pragma unroll
    for (int kc = 0; kc < 4; ++kc) {
#pragma unroll
        for (int e = 0; e < 16; ++e) {
            int k = kc * 32 + e + (hi << 4);
            a[kc][e] = (_Float16)Kmat[(m0 + mrow) * 128 + k];
        }
    }

    const int ntiles = (N + 15) >> 4;
    const int tstep  = gridDim.x << 4;   // columns advanced per grid stride
    for (int tile = blockIdx.x; tile < ntiles; tile += gridDim.x) {
        const int col0 = tile << 4;
        const bool full = (col0 + 16 <= N) && (BMODE != 0 || (NX & 3) == 0);
        const bool pfok = (col0 + tstep + 16 <= N);   // next tile fully valid

        __syncthreads();   // protect sB reuse (and first-iter stats init)
        if (full) {
            if (BMODE == 0) {
                // 512 float4 loads, row-major
                for (int t = threadIdx.x; t < 512; t += 256) {
                    const int k = t >> 2, n4 = (t & 3) << 2;
                    const float4 v4 = *(const float4*)&X[(size_t)k * NX + col0 + n4];
                    if (pfok)   // speculative prefetch of next grid-stride tile
                        __builtin_prefetch(&X[(size_t)k * NX + col0 + tstep + n4], 0, 3);
                    const int kc = k >> 5, kr = k & 31;
                    const int e = kr & 15, hb = (kr >> 4) << 4;
                    _Float16* bp = sB + kc * BCHUNK + e;
                    bp[(n4 + 0 + hb) * BSTRIDE] = (_Float16)v4.x;
                    bp[(n4 + 1 + hb) * BSTRIDE] = (_Float16)v4.y;
                    bp[(n4 + 2 + hb) * BSTRIDE] = (_Float16)v4.z;
                    bp[(n4 + 3 + hb) * BSTRIDE] = (_Float16)v4.w;
                }
            } else {
                // column-contiguous: 4 consecutive k -> one 8B LDS store
                for (int t = threadIdx.x; t < 512; t += 256) {
                    const int n = t >> 5, k4 = (t & 31) << 2;
                    const int col = col0 + n;
                    float4 v4;
                    if (BMODE == 2) {
                        const float4 va = *(const float4*)&X[(size_t)iInd[col] * 128 + k4];
                        const float4 vb = *(const float4*)&X[(size_t)jInd[col] * 128 + k4];
                        v4 = make_float4(va.x - vb.x, va.y - vb.y,
                                         va.z - vb.z, va.w - vb.w);
                        if (pfok) {   // prefetch next tile's indices
                            __builtin_prefetch(&iInd[col + tstep], 0, 3);
                            __builtin_prefetch(&jInd[col + tstep], 0, 3);
                        }
                    } else {
                        v4 = *(const float4*)&X[(size_t)col * 128 + k4];
                        if (pfok)
                            __builtin_prefetch(&X[(size_t)(col + tstep) * 128 + k4], 0, 3);
                    }
                    const int kc = k4 >> 5, kr = k4 & 31;
                    const int e = kr & 15, lslot = n + ((kr >> 4) << 4);
                    const v4h h4 = { (_Float16)v4.x, (_Float16)v4.y,
                                     (_Float16)v4.z, (_Float16)v4.w };
                    *(v4h*)(sB + kc * BCHUNK + lslot * BSTRIDE + e) = h4;
                }
            }
        } else {
            // partial tile: guarded scalar path (rare)
            for (int t = threadIdx.x; t < 2048; t += 256) {
                const int k = t >> 4, n = t & 15, col = col0 + n;
                float v = 0.0f;
                if (col < N) {
                    if (BMODE == 2)      v = X[(size_t)iInd[col] * 128 + k] -
                                             X[(size_t)jInd[col] * 128 + k];
                    else if (BMODE == 1) v = X[(size_t)col * 128 + k];
                    else                 v = X[(size_t)k * NX + col];
                }
                const int kc = k >> 5, kr = k & 31;
                sB[kc * BCHUNK + (n + ((kr >> 4) << 4)) * BSTRIDE + (kr & 15)] =
                    (_Float16)v;
            }
        }
        __syncthreads();

        // hoist all 4 B fragments (8x ds_load_b128) before the WMMA burst so
        // DS loads pipeline against matrix ops instead of wait-0 serializing
        union BU { v16h f; v8h h[2]; } bu[4];
#pragma unroll
        for (int kc = 0; kc < 4; ++kc) {
            const _Float16* bp = sB + kc * BCHUNK + lane * BSTRIDE;
            bu[kc].h[0] = *(const v8h*)(bp);        // ds_load_b128
            bu[kc].h[1] = *(const v8h*)(bp + 8);    // ds_load_b128
        }
        v8f c = {};
#pragma unroll
        for (int kc = 0; kc < 4; ++kc)
            c = __builtin_amdgcn_wmma_f32_16x16x32_f16(
                    false, a[kc], false, bu[kc].f, (short)0, c, false, false);

        const int col = col0 + mrow;
        float vout[8];
#pragma unroll
        for (int r = 0; r < 8; ++r)
            vout[r] = RELU ? fmaxf(c[r], 0.0f) : c[r];

        // independent per-row addresses (base + r*N) -> clusterable stores
        if (col < N) {
            float* __restrict__ yp = Y + (size_t)(m0 + (hi << 3)) * N + col;
#pragma unroll
            for (int r = 0; r < 8; ++r)
                yp[(size_t)r * (size_t)N] = vout[r];
        }
        if (STATS) {
#pragma unroll
            for (int r = 0; r < 8; ++r) {
                float s = (col < N) ? vout[r] : 0.0f;
                float q = s * s;
#pragma unroll
                for (int off = 1; off < 16; off <<= 1) {
                    s += __shfl_xor(s, off, 32);
                    q += __shfl_xor(q, off, 32);
                }
                if (mrow == 0) {   // lane 0 -> row m0+r, lane 16 -> row m0+8+r
                    const int row = m0 + r + (hi << 3);
                    sSum[row] += s;
                    sSq[row]  += q;
                }
            }
        }
    }

    if (STATS) {
        __syncthreads();
        if (threadIdx.x < 128) {
            atomicAdd(&gStats[2 * threadIdx.x],     sSum[threadIdx.x]);
            atomicAdd(&gStats[2 * threadIdx.x + 1], sSq[threadIdx.x]);
        }
    }
}

// zT[iInd[e]*128 + ch] += xe[ch, e];  zT[jInd[e]*128 + ch] -= xe[ch, e]
// Staged through LDS: coalesced xe reads, then 128-lane coalesced atomics
// (each edge endpoint touches 2 consecutive cachelines instead of 128 random).
#define SC_EDGES 64
__global__ void __launch_bounds__(256)
scatter_div(const float* __restrict__ xe, const int* __restrict__ iInd,
            const int* __restrict__ jInd, float* __restrict__ zT,
            int nE, int nN)
{
    __shared__ float sE[SC_EDGES * 130];   // pad 130 to dodge bank conflicts
    __shared__ int sI[SC_EDGES], sJ[SC_EDGES];
    const int e0 = blockIdx.x * SC_EDGES;
    const int t = threadIdx.x;
    if (t < SC_EDGES) {
        const int e = e0 + t;
        sI[t] = (e < nE) ? iInd[e] : 0;
    } else if (t < 2 * SC_EDGES) {
        const int e = e0 + t - SC_EDGES;
        sJ[t - SC_EDGES] = (e < nE) ? jInd[e] : 0;
    }
    const int le = t & 63;                 // edge within tile
    const int cb = t >> 6;                 // 4 channels per pass
    const int eg = e0 + le;
#pragma unroll 4
    for (int ch = cb; ch < 128; ch += 4)
        sE[le * 130 + ch] = (eg < nE) ? xe[(size_t)ch * nE + eg] : 0.0f;
    __syncthreads();

    const int ch = t & 127;
    const int eo = t >> 7;                 // 2 edges per pass
    for (int le2 = eo; le2 < SC_EDGES; le2 += 2) {
        const int e = e0 + le2;
        if (e < nE) {
            const float v = sE[le2 * 130 + ch];
            atomicAdd(&zT[(size_t)sI[le2] * 128 + ch],  v);
            atomicAdd(&zT[(size_t)sJ[le2] * 128 + ch], -v);
        }
    }
}

// Xio += H * relu((Y - mean_row) * rsqrt(var_row + eps)), stats from gStats
__global__ void scale_update(float* __restrict__ Xio, const float* __restrict__ Y,
                             const float* __restrict__ gStats, int N)
{
    const int row  = blockIdx.y;
    const float s  = gStats[2 * row];
    const float q  = gStats[2 * row + 1];
    const float ni = 1.0f / (float)N;
    const float mean = s * ni;
    const float var  = q - s * s * ni;       // sum of centered squares
    const float inv  = rsqrtf(var + TVEPS);

    const int gid = blockIdx.x * blockDim.x + threadIdx.x;
    const int stride = gridDim.x * blockDim.x;
    float* __restrict__ xr = Xio + (size_t)row * N;
    const float* __restrict__ yr = Y + (size_t)row * N;

    if ((N & 3) == 0) {
        float4* x4 = (float4*)xr;
        const float4* y4 = (const float4*)yr;
        const int n4 = N >> 2;
        for (int c = gid; c < n4; c += stride) {
            float4 y = y4[c];
            float4 x = x4[c];
            x.x += H_STEP * fmaxf((y.x - mean) * inv, 0.0f);
            x.y += H_STEP * fmaxf((y.y - mean) * inv, 0.0f);
            x.z += H_STEP * fmaxf((y.z - mean) * inv, 0.0f);
            x.w += H_STEP * fmaxf((y.w - mean) * inv, 0.0f);
            x4[c] = x;
        }
    } else {
        for (int c = gid; c < N; c += stride) {
            const float v = (yr[c] - mean) * inv;
            xr[c] += H_STEP * fmaxf(v, 0.0f);
        }
    }
}

extern "C" void kernel_launch(void* const* d_in, const int* in_sizes, int n_in,
                              void* d_out, int out_size, void* d_ws, size_t ws_size,
                              hipStream_t stream)
{
    const float* xn      = (const float*)d_in[0];
    const float* xe      = (const float*)d_in[1];
    const int*   iInd    = (const int*)d_in[2];
    const int*   jInd    = (const int*)d_in[3];
    const float* KNopen  = (const float*)d_in[4];
    const float* KEopen  = (const float*)d_in[5];
    const float* KNclose = (const float*)d_in[6];
    // d_in[7] = KEclose: unused (reference applies KNclose to xe as well)
    const float* KN      = (const float*)d_in[8];   // [4,128,128]
    const float* KE      = (const float*)d_in[9];   // [4,128,128]

    const int nN = in_sizes[0] / 128;
    const int nE = in_sizes[2];

    float* ws   = (float*)d_ws;
    float* xn2  = ws;  ws += (size_t)128 * nN;
    float* xn2T = ws;  ws += (size_t)128 * nN;   // [nN, 128]
    float* xe2  = ws;  ws += (size_t)128 * nE;
    float* Ye   = ws;  ws += (size_t)128 * nE;
    float* zT   = ws;  ws += (size_t)128 * nN;   // [nN, 128]
    float* Yn   = ws;  ws += (size_t)128 * nN;
    float* stE  = ws;  ws += 256;
    float* stN  = ws;  ws += 256;

    float* out_xn = (float*)d_out;
    float* out_xe = out_xn + (size_t)128 * nN;

    const dim3 blk(256);
    const int gN = imin_host((nN + 15) / 16, 1024);
    const int gE = imin_host((nE + 15) / 16, 2048);
    const dim3 gT((nN + 31) / 32, 4);
    const dim3 gUpdN(imin_host((nN / 4 + 255) / 256, 1024), 128);
    const dim3 gUpdE(imin_host((nE / 4 + 255) / 256, 2048), 128);
    const dim3 gScat((nE + SC_EDGES - 1) / SC_EDGES);

    // opens (row-major B)
    gemm_k128<0, false, false><<<gN, blk, 0, stream>>>(KNopen, xn, nullptr, nullptr, xn2, nullptr, nN, nN);
    gemm_k128<0, false, false><<<gE, blk, 0, stream>>>(KEopen, xe, nullptr, nullptr, xe2, nullptr, nE, nE);

    for (int i = 0; i < 4; ++i) {
        const float* KNi = KN + (size_t)i * 128 * 128;
        const float* KEi = KE + (size_t)i * 128 * 128;

        // xn2T = xn2^T  (makes gathered columns contiguous; L2-resident)
        transpose128<<<gT, blk, 0, stream>>>(xn2, xn2T, nN);

        // Ai = KN[i] @ (xn2[:,i] - xn2[:,j]) with fused row stats
        zero_kernel<<<1, 256, 0, stream>>>(stE, 256);
        gemm_k128<2, false, true><<<gE, blk, 0, stream>>>(KNi, xn2T, iInd, jInd, Ye, stE, nE, nN);
        // xe2 += H * relu(tv_norm(Ai))
        scale_update<<<gUpdE, blk, 0, stream>>>(xe2, Ye, stE, nE);

        // zT = edge_div(xe2)^T  (coalesced atomics)
        zero_kernel<<<2048, 256, 0, stream>>>(zT, 128 * nN);
        scatter_div<<<gScat, blk, 0, stream>>>(xe2, iInd, jInd, zT, nE, nN);

        // Bi' = relu(KE[i] @ z) with fused row stats (column-contiguous B)
        zero_kernel<<<1, 256, 0, stream>>>(stN, 256);
        gemm_k128<1, true, true><<<gN, blk, 0, stream>>>(KEi, zT, nullptr, nullptr, Yn, stN, nN, nN);
        // xn2 += H * relu(tv_norm(Bi'))
        scale_update<<<gUpdN, blk, 0, stream>>>(xn2, Yn, stN, nN);
    }

    // closes (reference uses KNclose for both outputs)
    gemm_k128<0, false, false><<<gN, blk, 0, stream>>>(KNclose, xn2, nullptr, nullptr, out_xn, nullptr, nN, nN);
    gemm_k128<0, false, false><<<gE, blk, 0, stream>>>(KNclose, xe2, nullptr, nullptr, out_xe, nullptr, nE, nE);
}
